// MHABlock_89653147337194
// MI455X (gfx1250) — matrix operand order repaired and checked by reading, not compiled, not run on hardware
//
#include <hip/hip_runtime.h>
#include <hip/hip_bf16.h>
#include <math.h>

// ---------------------------------------------------------------------------
// Transformer block (pre-norm causal MHA + GELU FFN) for gfx1250 (MI455X).
// f16 WMMA (v_wmma_f32_16x16x32_f16) everywhere, f32 epilogues.
// Tile staging via the Tensor Data Mover (tensor_load_to_lds + TENSORcnt)
// when the builtin is available; falls back to b128 cooperative copies.
// ---------------------------------------------------------------------------

typedef __attribute__((ext_vector_type(16))) _Float16 v16h;
typedef __attribute__((ext_vector_type(8)))  float    v8f;

#define BB 64
#define TT 256
#define CCH 384
#define HH 6
#define HSZ 64
#define NROWS (BB * TT)      // 16384

#if defined(__AMDGCN__) && __has_builtin(__builtin_amdgcn_tensor_load_to_lds) && \
    __has_builtin(__builtin_amdgcn_s_wait_tensorcnt)
#define USE_TDM 1
#else
#define USE_TDM 0
#endif

#if USE_TDM
typedef unsigned int tdm_v4u __attribute__((ext_vector_type(4)));
typedef int          tdm_v8i __attribute__((ext_vector_type(8)));
typedef int          tdm_v4i __attribute__((ext_vector_type(4)));

// low 32 bits of a flat LDS pointer are the workgroup-relative LDS byte addr
static __device__ __forceinline__ unsigned lds_addr_of(const void* p) {
  return (unsigned)(unsigned long long)p;
}

#define TD_BIG 0x100000u

// 2D tile load, 2-byte elements. pad_i: 2^(code+1) DWORDs between pads;
// pad_a: (code+1) DWORDs of pad. Groups 2/3 (and trailing group) zero (<=2D).
static __device__ __forceinline__ void tdm_load_2d(
    const void* gaddr, unsigned lds_addr, unsigned tensor_d0,
    unsigned tensor_d1, unsigned stride0, unsigned tile_d0, unsigned tile_d1,
    unsigned pad_i, unsigned pad_a) {
  unsigned long long ga = (unsigned long long)gaddr;
  tdm_v4u g0;
  g0[0] = 1u;                                              // count=1 (valid)
  g0[1] = lds_addr;                                        // lds_addr
  g0[2] = (unsigned)ga;                                    // global_addr lo
  g0[3] = (unsigned)((ga >> 32) & 0x01FFFFFFull) | (2u << 30);  // hi | type=2
  tdm_v8i g1;
  g1[0] = (int)((1u << 16) | (1u << 20) | (pad_i << 22) | (pad_a << 25));
  g1[1] = (int)((tensor_d0 & 0xFFFFu) << 16);
  g1[2] = (int)(((tensor_d0 >> 16) & 0xFFFFu) | ((tensor_d1 & 0xFFFFu) << 16));
  g1[3] = (int)(((tensor_d1 >> 16) & 0xFFFFu) | ((tile_d0 & 0xFFFFu) << 16));
  g1[4] = (int)(tile_d1 & 0xFFFFu);
  g1[5] = (int)stride0;
  g1[6] = 0;
  g1[7] = 0;
  tdm_v4i gz4 = {0, 0, 0, 0};
  tdm_v8i gz8 = {0, 0, 0, 0, 0, 0, 0, 0};
  __builtin_amdgcn_tensor_load_to_lds(g0, g1, gz4, gz4, gz8, 0);
}
#endif  // USE_TDM

// ---------------- WMMA fragment loaders (CDNA5 ISA layouts) ----------------
// A (16x32 f16, MxK): lane L holds row M=L&15; element e -> K = 16*(e>>3)+8*(L>>4)+(e&7)
static __device__ __forceinline__ v16h wmma_load_a(const _Float16* src, int ld,
                                                   int m0, int k0, int lane) {
  v16h a;
  const int m  = m0 + (lane & 15);
  const int kb = k0 + ((lane >> 4) << 3);
#pragma unroll
  for (int e = 0; e < 16; ++e) {
    int k = kb + ((e >> 3) << 4) + (e & 7);
    a[e] = src[(size_t)m * ld + k];
  }
  return a;
}

// B (32x16, KxN) where the source is row-major [N][K] (B = src^T):
// lane L holds col N=L&15; element e -> K = 16*(e>>3)+8*(L>>4)+(e&7)
static __device__ __forceinline__ v16h wmma_load_bt(const _Float16* src, int ld,
                                                    int k0, int n0, int lane) {
  v16h b;
  const int n  = n0 + (lane & 15);
  const int kb = k0 + ((lane >> 4) << 3);
#pragma unroll
  for (int e = 0; e < 16; ++e)
    b[e] = src[(size_t)n * ld + kb + ((e >> 3) << 4) + (e & 7)];
  return b;
}

// ---------------------------- small prep kernels ---------------------------
__global__ void pack_qkv_kernel(const float* __restrict__ Wq,
                                const float* __restrict__ Wk,
                                const float* __restrict__ Wv,
                                _Float16* __restrict__ Wqkv_t) {
  int i = blockIdx.x * 256 + threadIdx.x;           // 3*H*C*HS = 442368
  if (i >= 3 * HH * CCH * HSZ) return;
  int sel  = i / (HH * CCH * HSZ);
  int rem  = i - sel * (HH * CCH * HSZ);
  int hh   = rem / (CCH * HSZ);
  int rem2 = rem - hh * (CCH * HSZ);
  int c    = rem2 / HSZ;
  int d    = rem2 - c * HSZ;
  const float* W = (sel == 0) ? Wq : (sel == 1) ? Wk : Wv;
  Wqkv_t[(size_t)(sel * CCH + hh * HSZ + d) * CCH + c] =
      (_Float16)W[(size_t)(hh * CCH + c) * HSZ + d];
}

__global__ void transpose_f32_to_f16_kernel(const float* __restrict__ in,
                                            _Float16* __restrict__ out,
                                            int Kdim, int Ndim) {
  int i = blockIdx.x * 256 + threadIdx.x;
  if (i >= Kdim * Ndim) return;
  int k = i / Ndim, n = i - k * Ndim;
  out[(size_t)n * Kdim + k] = (_Float16)in[i];
}

__global__ __launch_bounds__(128) void ln_kernel(const float* __restrict__ x,
                                                 const float* __restrict__ w,
                                                 const float* __restrict__ bb,
                                                 _Float16* __restrict__ out) {
  const int row = blockIdx.x;
  const int tid = threadIdx.x;
  const float* xr = x + (size_t)row * CCH;
  float a0 = xr[tid], a1 = xr[tid + 128], a2 = xr[tid + 256];
  __shared__ float red[128];
  red[tid] = a0 + a1 + a2;
  __syncthreads();
#pragma unroll
  for (int off = 64; off > 0; off >>= 1) {
    if (tid < off) red[tid] += red[tid + off];
    __syncthreads();
  }
  float mean = red[0] * (1.0f / CCH);
  __syncthreads();
  float d0 = a0 - mean, d1 = a1 - mean, d2 = a2 - mean;
  red[tid] = d0 * d0 + d1 * d1 + d2 * d2;
  __syncthreads();
#pragma unroll
  for (int off = 64; off > 0; off >>= 1) {
    if (tid < off) red[tid] += red[tid + off];
    __syncthreads();
  }
  float rstd = rsqrtf(red[0] * (1.0f / CCH) + 1e-5f);
  _Float16* orow = out + (size_t)row * CCH;
  orow[tid]       = (_Float16)(d0 * rstd * w[tid]       + bb[tid]);
  orow[tid + 128] = (_Float16)(d1 * rstd * w[tid + 128] + bb[tid + 128]);
  orow[tid + 256] = (_Float16)(d2 * rstd * w[tid + 256] + bb[tid + 256]);
}

// ------------------------------ WMMA GEMM ----------------------------------
// C[M,N] = A[M,K](f16 row-major) * Bt[N,K](f16, transposed weights).
// 128x64 block tile, 4 waves, 8 wmma/wave/k-step, LDS double-buffered.
// TDM path: wave 0 DMAs next A/B tiles into the idle buffer during compute.
#define LDA  40   // 32 + 8 halfs padding (16 + 4 DWORDs -> pad codes 3/3)
#define LDBK 40

template <int MODE>
__global__ __launch_bounds__(128) void gemm_kernel(
    const _Float16* __restrict__ A, const _Float16* __restrict__ Bt,
    int M, int Ncols, int K,
    const float* __restrict__ bias, const float* __restrict__ resid,
    float* __restrict__ outf, _Float16* __restrict__ outh,
    _Float16* __restrict__ Qp, _Float16* __restrict__ Kp,
    _Float16* __restrict__ Vp) {
  __shared__ __align__(16) _Float16 As[2][128 * LDA];
  __shared__ __align__(16) _Float16 Bs[2][64 * LDBK];
  const int tid  = threadIdx.x;
  const int lane = tid & 31;
  const int w    = tid >> 5;
  const int m0   = blockIdx.y * 128;
  const int n0   = blockIdx.x * 64;

  v8f acc[8] = {};
  const int nit = K / 32;

#if USE_TDM
  if (w == 0) {
    tdm_load_2d(&A[(size_t)m0 * K], lds_addr_of(&As[0][0]), TD_BIG, TD_BIG,
                (unsigned)K, 32, 128, 3, 3);
    tdm_load_2d(&Bt[(size_t)n0 * K], lds_addr_of(&Bs[0][0]), TD_BIG, TD_BIG,
                (unsigned)K, 32, 64, 3, 3);
    __builtin_amdgcn_s_wait_tensorcnt((unsigned short)0);
  }
  __syncthreads();
  for (int it = 0; it < nit; ++it) {
    const int cur = it & 1;
    const bool has_next = (it + 1) < nit;
    if (has_next && w == 0) {
      const int kn = (it + 1) * 32;
      tdm_load_2d(&A[(size_t)m0 * K + kn], lds_addr_of(&As[1 - cur][0]),
                  TD_BIG, TD_BIG, (unsigned)K, 32, 128, 3, 3);
      tdm_load_2d(&Bt[(size_t)n0 * K + kn], lds_addr_of(&Bs[1 - cur][0]),
                  TD_BIG, TD_BIG, (unsigned)K, 32, 64, 3, 3);
    }
    v16h bfrag = wmma_load_bt(&Bs[cur][0], LDBK, 0, w * 16, lane);
#pragma unroll
    for (int mt = 0; mt < 8; ++mt) {
      v16h afrag = wmma_load_a(&As[cur][0], LDA, mt * 16, 0, lane);
      acc[mt] = __builtin_amdgcn_wmma_f32_16x16x32_f16(
          false, afrag, false, bfrag, (short)0, acc[mt], false, false);
    }
    if (has_next && w == 0)
      __builtin_amdgcn_s_wait_tensorcnt((unsigned short)0);
    __syncthreads();
  }
#else
  const int arow[4] = {(tid * 4 + 0) >> 2, (tid * 4 + 1) >> 2,
                       (tid * 4 + 2) >> 2, (tid * 4 + 3) >> 2};
  const int acol[4] = {((tid * 4 + 0) & 3) << 3, ((tid * 4 + 1) & 3) << 3,
                       ((tid * 4 + 2) & 3) << 3, ((tid * 4 + 3) & 3) << 3};
  const int brow[2] = {(tid * 2 + 0) >> 2, (tid * 2 + 1) >> 2};
  const int bcol[2] = {((tid * 2 + 0) & 3) << 3, ((tid * 2 + 1) & 3) << 3};
#pragma unroll
  for (int i = 0; i < 4; ++i)
    *(uint4*)&As[0][arow[i] * LDA + acol[i]] =
        *(const uint4*)&A[(size_t)(m0 + arow[i]) * K + acol[i]];
#pragma unroll
  for (int i = 0; i < 2; ++i)
    *(uint4*)&Bs[0][brow[i] * LDBK + bcol[i]] =
        *(const uint4*)&Bt[(size_t)(n0 + brow[i]) * K + bcol[i]];
  __syncthreads();
  for (int it = 0; it < nit; ++it) {
    const int cur = it & 1;
    const bool has_next = (it + 1) < nit;
    uint4 pa[4], pb[2];
    if (has_next) {
      const int kn = (it + 1) * 32;
#pragma unroll
      for (int i = 0; i < 4; ++i)
        pa[i] = *(const uint4*)&A[(size_t)(m0 + arow[i]) * K + kn + acol[i]];
#pragma unroll
      for (int i = 0; i < 2; ++i)
        pb[i] = *(const uint4*)&Bt[(size_t)(n0 + brow[i]) * K + kn + bcol[i]];
    }
    v16h bfrag = wmma_load_bt(&Bs[cur][0], LDBK, 0, w * 16, lane);
#pragma unroll
    for (int mt = 0; mt < 8; ++mt) {
      v16h afrag = wmma_load_a(&As[cur][0], LDA, mt * 16, 0, lane);
      acc[mt] = __builtin_amdgcn_wmma_f32_16x16x32_f16(
          false, afrag, false, bfrag, (short)0, acc[mt], false, false);
    }
    if (has_next) {
      const int nxt = 1 - cur;
#pragma unroll
      for (int i = 0; i < 4; ++i)
        *(uint4*)&As[nxt][arow[i] * LDA + acol[i]] = pa[i];
#pragma unroll
      for (int i = 0; i < 2; ++i)
        *(uint4*)&Bs[nxt][brow[i] * LDBK + bcol[i]] = pb[i];
    }
    __syncthreads();
  }
#endif

  const int gn    = n0 + w * 16 + (lane & 15);
  const int mbase = m0 + ((lane >> 4) << 3);
  const float bv  = bias[gn];
#pragma unroll
  for (int mt = 0; mt < 8; ++mt) {
#pragma unroll
    for (int r = 0; r < 8; ++r) {
      int gm  = mbase + mt * 16 + r;
      float v = acc[mt][r] + bv;
      if (MODE == 0) {
        int sel = (gn >= 768) ? 2 : (gn >= 384) ? 1 : 0;
        int cc  = gn - sel * CCH;
        int hh = cc >> 6, d = cc & 63;
        int bidx = gm >> 8, t = gm & 255;
        if (sel == 2)  // V transposed: [b,h][d][t]
          Vp[((size_t)(bidx * HH + hh) * HSZ + d) * TT + t] = (_Float16)v;
        else {
          _Float16* dst = (sel == 0) ? Qp : Kp;
          dst[(((size_t)(bidx * HH + hh)) * TT + t) * HSZ + d] = (_Float16)v;
        }
      } else if (MODE == 1 || MODE == 3) {
        size_t idx = (size_t)gm * Ncols + gn;
        outf[idx] = v + resid[idx];
      } else {  // MODE 2: exact (erf) GELU
        float g = 0.5f * v * (1.0f + erff(v * 0.70710678118654752f));
        outh[(size_t)gm * Ncols + gn] = (_Float16)g;
      }
    }
  }
}

// --------------------------- causal flash attention ------------------------
// grid = B*H*(T/64); block = 128 (4 waves). K stored [t][d], V stored
// transposed [d][t]. TDM stages the 64x64 K / V^T tiles (pad 32+4 DWORDs
// matching LDK=72) while softmax state lives in registers.
#define LDK 72

__global__ __launch_bounds__(128) void attn_kernel(
    const _Float16* __restrict__ Q, const _Float16* __restrict__ Kt,
    const _Float16* __restrict__ V, _Float16* __restrict__ O) {
  __shared__ __align__(16) _Float16 Ks[64 * LDK];  // [key t][d]
  __shared__ __align__(16) _Float16 Vs[64 * LDK];  // [d][key t]
  __shared__ __align__(16) _Float16 Ps[64 * LDK];  // per-wave 16-row strips

  const int tid  = threadIdx.x;
  const int lane = tid & 31;
  const int w    = tid >> 5;
  const int tq   = blockIdx.x & 3;
  const int bh   = blockIdx.x >> 2;
  const int h    = bh % HH;
  const int b    = bh / HH;

  const _Float16* Qb  = Q  + (size_t)bh * TT * HSZ;  // [t][d]
  const _Float16* Kb  = Kt + (size_t)bh * TT * HSZ;  // [t][d]
  const _Float16* Vtb = V  + (size_t)bh * HSZ * TT;  // [d][t]

  const int mrow = tq * 64 + w * 16;
  float m_run[8], l_run[8];
  v8f o_acc[4] = {};
#pragma unroll
  for (int r = 0; r < 8; ++r) { m_run[r] = -3.0e38f; l_run[r] = 0.0f; }

  // Q fragments are invariant over key blocks: hoist
  v16h aq[2];
#pragma unroll
  for (int kk = 0; kk < 2; ++kk)
    aq[kk] = wmma_load_a(Qb, HSZ, mrow, kk * 32, lane);

  for (int j = 0; j <= tq; ++j) {
    __syncthreads();
#if USE_TDM
    if (w == 0) {
      tdm_load_2d(Kb + (size_t)j * 64 * HSZ, lds_addr_of(Ks), TD_BIG, TD_BIG,
                  HSZ, 64, 64, 4, 3);
      tdm_load_2d(Vtb + j * 64, lds_addr_of(Vs), TD_BIG, TD_BIG,
                  TT, 64, 64, 4, 3);
      __builtin_amdgcn_s_wait_tensorcnt((unsigned short)0);
    }
#else
#pragma unroll
    for (int i = 0; i < 4; ++i) {
      int c = tid * 4 + i;
      int row = c >> 3, col = (c & 7) << 3;
      *(uint4*)&Ks[row * LDK + col] =
          *(const uint4*)&Kb[(size_t)(j * 64 + row) * HSZ + col];
      *(uint4*)&Vs[row * LDK + col] =
          *(const uint4*)&Vtb[(size_t)row * TT + j * 64 + col];
    }
#endif
    __syncthreads();

    // S(16x64) = Q(16x64) * K^T
    v8f s[4] = {};
#pragma unroll
    for (int kk = 0; kk < 2; ++kk) {
#pragma unroll
      for (int t = 0; t < 4; ++t) {
        v16h bfr = wmma_load_bt(Ks, LDK, kk * 32, t * 16, lane);
        s[t] = __builtin_amdgcn_wmma_f32_16x16x32_f16(
            false, aq[kk], false, bfr, (short)0, s[t], false, false);
      }
    }

    // scale + causal mask (diagonal block only)
#pragma unroll
    for (int t = 0; t < 4; ++t) {
#pragma unroll
      for (int r = 0; r < 8; ++r) {
        float vv = s[t][r] * 0.125f;  // 1/sqrt(64)
        if (j == tq) {
          int n_g = j * 64 + t * 16 + (lane & 15);
          int m_g = mrow + r + ((lane >> 4) << 3);
          if (n_g > m_g) vv = -3.0e38f;
        }
        s[t][r] = vv;
      }
    }

    // online softmax (rows live in the D layout: row = r + 8*(lane>=16))
    float mb[8];
#pragma unroll
    for (int r = 0; r < 8; ++r)
      mb[r] = fmaxf(fmaxf(s[0][r], s[1][r]), fmaxf(s[2][r], s[3][r]));
#pragma unroll
    for (int off = 1; off < 16; off <<= 1)
#pragma unroll
      for (int r = 0; r < 8; ++r)
        mb[r] = fmaxf(mb[r], __shfl_xor(mb[r], off, 32));

    float scale[8], ls[8];
#pragma unroll
    for (int r = 0; r < 8; ++r) {
      float mn = fmaxf(m_run[r], mb[r]);
      scale[r] = __expf(m_run[r] - mn);
      m_run[r] = mn;
      ls[r] = 0.0f;
    }
#pragma unroll
    for (int t = 0; t < 4; ++t)
#pragma unroll
      for (int r = 0; r < 8; ++r) {
        float pp = __expf(s[t][r] - m_run[r]);
        s[t][r] = pp;
        ls[r] += pp;
      }
#pragma unroll
    for (int off = 1; off < 16; off <<= 1)
#pragma unroll
      for (int r = 0; r < 8; ++r) ls[r] += __shfl_xor(ls[r], off, 32);
#pragma unroll
    for (int r = 0; r < 8; ++r) l_run[r] = l_run[r] * scale[r] + ls[r];
#pragma unroll
    for (int t = 0; t < 4; ++t)
#pragma unroll
      for (int r = 0; r < 8; ++r) o_acc[t][r] *= scale[r];

    // P (f16) -> this wave's private LDS strip (D-layout -> A-layout reshape)
    _Float16* Pw = Ps + (size_t)w * 16 * LDK;
#pragma unroll
    for (int t = 0; t < 4; ++t)
#pragma unroll
      for (int r = 0; r < 8; ++r)
        Pw[(size_t)(r + ((lane >> 4) << 3)) * LDK + t * 16 + (lane & 15)] =
            (_Float16)s[t][r];

    // O(16x64) += P(16x64) * V(64x64)
#pragma unroll
    for (int kk = 0; kk < 2; ++kk) {
      v16h a = wmma_load_a(Pw, LDK, 0, kk * 32, lane);
#pragma unroll
      for (int t = 0; t < 4; ++t) {
        v16h bfr = wmma_load_bt(Vs, LDK, kk * 32, t * 16, lane);
        o_acc[t] = __builtin_amdgcn_wmma_f32_16x16x32_f16(
            false, a, false, bfr, (short)0, o_acc[t], false, false);
      }
    }
  }

  // normalize + store concat-head f16
#pragma unroll
  for (int t = 0; t < 4; ++t)
#pragma unroll
    for (int r = 0; r < 8; ++r) {
      int tg  = mrow + r + ((lane >> 4) << 3);
      int col = h * HSZ + t * 16 + (lane & 15);
      float o = o_acc[t][r] / l_run[r];
      O[(size_t)(b * TT + tg) * CCH + col] = (_Float16)o;
    }
}

// ------------------------------- launcher ----------------------------------
extern "C" void kernel_launch(void* const* d_in, const int* in_sizes, int n_in,
                              void* d_out, int out_size, void* d_ws,
                              size_t ws_size, hipStream_t stream) {
  (void)in_sizes; (void)n_in; (void)out_size; (void)ws_size;
  const float* x     = (const float*)d_in[0];
  const float* ln1w  = (const float*)d_in[1];
  const float* ln1b  = (const float*)d_in[2];
  const float* Wq    = (const float*)d_in[3];
  const float* bq    = (const float*)d_in[4];
  const float* Wk    = (const float*)d_in[5];
  const float* bk    = (const float*)d_in[6];
  const float* Wv    = (const float*)d_in[7];
  const float* bv    = (const float*)d_in[8];
  const float* Wproj = (const float*)d_in[9];
  const float* bproj = (const float*)d_in[10];
  const float* ln2w  = (const float*)d_in[11];
  const float* ln2b  = (const float*)d_in[12];
  const float* W1    = (const float*)d_in[13];
  const float* b1    = (const float*)d_in[14];
  const float* W2    = (const float*)d_in[15];
  const float* b2    = (const float*)d_in[16];

  char* p = (char*)d_ws;
  auto wsalloc = [&](size_t bytes) -> char* {
    char* r = p;
    p += (bytes + 255) & ~(size_t)255;
    return r;
  };
  _Float16* Wqkv16  = (_Float16*)wsalloc((size_t)CCH * 1152 * 2);  // [1152][384]
  float*    qkvBias = (float*)   wsalloc(1152 * 4);
  _Float16* Wproj16 = (_Float16*)wsalloc((size_t)CCH * CCH * 2);
  _Float16* W116    = (_Float16*)wsalloc((size_t)CCH * 4 * CCH * 2);
  _Float16* W216    = (_Float16*)wsalloc((size_t)4 * CCH * CCH * 2);
  _Float16* Hbuf    = (_Float16*)wsalloc((size_t)NROWS * CCH * 2);
  _Float16* Qp      = (_Float16*)wsalloc((size_t)NROWS * CCH * 2);
  _Float16* Kp      = (_Float16*)wsalloc((size_t)NROWS * CCH * 2);
  _Float16* Vp      = (_Float16*)wsalloc((size_t)NROWS * CCH * 2);
  _Float16* Op      = (_Float16*)wsalloc((size_t)NROWS * CCH * 2);
  float*    X1      = (float*)   wsalloc((size_t)NROWS * CCH * 4);
  _Float16* Fbuf    = Qp;  // reuse Q..O region for [16384,1536] f16 (exact fit)

  pack_qkv_kernel<<<(3 * HH * CCH * HSZ + 255) / 256, 256, 0, stream>>>(
      Wq, Wk, Wv, Wqkv16);
  (void)hipMemcpyAsync(qkvBias,       bq, CCH * 4, hipMemcpyDeviceToDevice, stream);
  (void)hipMemcpyAsync(qkvBias + 384, bk, CCH * 4, hipMemcpyDeviceToDevice, stream);
  (void)hipMemcpyAsync(qkvBias + 768, bv, CCH * 4, hipMemcpyDeviceToDevice, stream);
  transpose_f32_to_f16_kernel<<<(CCH * CCH + 255) / 256, 256, 0, stream>>>(
      Wproj, Wproj16, CCH, CCH);
  transpose_f32_to_f16_kernel<<<(CCH * 4 * CCH + 255) / 256, 256, 0, stream>>>(
      W1, W116, CCH, 4 * CCH);
  transpose_f32_to_f16_kernel<<<(4 * CCH * CCH + 255) / 256, 256, 0, stream>>>(
      W2, W216, 4 * CCH, CCH);

  ln_kernel<<<NROWS, 128, 0, stream>>>(x, ln1w, ln1b, Hbuf);
  gemm_kernel<0><<<dim3(1152 / 64, NROWS / 128), 128, 0, stream>>>(
      Hbuf, Wqkv16, NROWS, 1152, CCH, qkvBias, nullptr, nullptr, nullptr,
      Qp, Kp, Vp);
  attn_kernel<<<BB * HH * (TT / 64), 128, 0, stream>>>(Qp, Kp, Vp, Op);
  gemm_kernel<1><<<dim3(CCH / 64, NROWS / 128), 128, 0, stream>>>(
      Op, Wproj16, NROWS, CCH, CCH, bproj, x, X1, nullptr,
      nullptr, nullptr, nullptr);
  ln_kernel<<<NROWS, 128, 0, stream>>>(X1, ln2w, ln2b, Hbuf);
  gemm_kernel<2><<<dim3(4 * CCH / 64, NROWS / 128), 128, 0, stream>>>(
      Hbuf, W116, NROWS, 4 * CCH, CCH, b1, nullptr, nullptr, Fbuf,
      nullptr, nullptr, nullptr);
  gemm_kernel<3><<<dim3(CCH / 64, NROWS / 128), 128, 0, stream>>>(
      Fbuf, W216, NROWS, CCH, 4 * CCH, b2, X1, (float*)d_out, nullptr,
      nullptr, nullptr, nullptr);
}
